// AL_ARCF_68659347194112
// MI455X (gfx1250) — matrix operand order repaired and checked by reading, not compiled, not run on hardware
//
#include <hip/hip_runtime.h>
#include <hip/hip_bf16.h>

// ---------------------------------------------------------------------------
// AL_ARCF fusion forward for MI455X (gfx1250, wave32, WMMA).
//
// Algebraic simplifications (exact, see analysis):
//   * m2 == false always (joint and delta are the identical expression)
//   * m1 == false always (three values cannot all strictly exceed their mean)
//       => fusion == f_cross;  f_concat / f_hybrid / cosines are dead code.
//   * einsum('bhqk,bhvd->bhqd') sums attn over k (softmax rows == 1) and v
//     over seq independently => cross-attn == broadcast of Sum_s V[s];
//     Q/K projections and softmax are dead code, and
//       f_cross[b,:] = (visSum@vW0^T + acSum@(vW1+vW2)^T + 50*(vb0+vb1+vb2))/3
// Remaining heavy compute: the LSTM recurrence -> 50 sequential
// [512 x 3072] = [512 x 1536] @ [1536 x 3072]^T WMMA GEMM steps (f16 in,
// f32 accumulate via v_wmma_f32_16x16x32_f16).
// ---------------------------------------------------------------------------

#define D     768
#define N_SEQ 50
#define B_SZ  512
#define G4    (4*D)          // 3072
#define FLAT  (N_SEQ*D)      // 38400

typedef __attribute__((ext_vector_type(8)))  _Float16 v8h;
typedef __attribute__((ext_vector_type(16))) _Float16 v16h;
typedef __attribute__((ext_vector_type(8)))  float    v8f;

__device__ __forceinline__ v16h cat16(v8h lo, v8h hi) {
  return __builtin_shufflevector(lo, hi, 0,1,2,3,4,5,6,7,8,9,10,11,12,13,14,15);
}
__device__ __forceinline__ float sigm(float x) { return 1.f / (1.f + __expf(-x)); }

// ---------------------------------------------------------------------------
// EGCE: gap over the 50 flat elements of each of 768 "channels" (the faithful
// reshape means flat index f == ch*50+w == n*768+c), k=3 conv over channels,
// sigmoid gate, multiply. For text we materialize the gated tensor (needed by
// the LN residual and the final residual). For visual/acoustic we only need
// per-batch column sums (Sum over n) for the collapsed cross-attention.
// blockIdx.x = batch, blockIdx.y = stream (0=text,1=visual,2=acoustic)
// ---------------------------------------------------------------------------
__global__ __launch_bounds__(256)
void egce_kernel(const float* __restrict__ te, const int* __restrict__ vid,
                 const int* __restrict__ aid, const float* __restrict__ vemb,
                 const float* __restrict__ aemb, const float* __restrict__ w3,
                 float* __restrict__ textG, float* __restrict__ visSum,
                 float* __restrict__ acSum)
{
  const int b = blockIdx.x, s = blockIdx.y, tid = threadIdx.x;
  __shared__ float gap[D];
  __shared__ float gate[D];

  auto val = [&](int f) -> float {
    int n = f / D, c = f - n * D;
    if (s == 0) return te[(size_t)b * FLAT + f];
    if (s == 1) { float v = vemb[vid[b * N_SEQ + n] * D + c]; return v > 0.f ? v : 0.f; }
    return aemb[aid[b * N_SEQ + n] * D + c];
  };

  for (int ch = tid; ch < D; ch += 256) {
    float acc = 0.f;
    const int f0 = ch * N_SEQ;
    for (int w = 0; w < N_SEQ; ++w) acc += val(f0 + w);
    gap[ch] = acc * (1.f / N_SEQ);
  }
  __syncthreads();
  const float w0 = w3[0], w1 = w3[1], w2 = w3[2];
  for (int ch = tid; ch < D; ch += 256) {
    float l = ch ? gap[ch - 1] : 0.f;
    float r = (ch < D - 1) ? gap[ch + 1] : 0.f;
    gate[ch] = sigm(w0 * l + w1 * gap[ch] + w2 * r);
  }
  __syncthreads();
  if (s == 0) {
    for (int f = tid; f < FLAT; f += 256)
      textG[(size_t)b * FLAT + f] = val(f) * gate[f / N_SEQ];
  } else {
    float* dst = (s == 1) ? visSum : acSum;
    for (int c = tid; c < D; c += 256) {
      float acc = 0.f;
      for (int n = 0; n < N_SEQ; ++n) {
        int f = n * D + c;
        acc += val(f) * gate[f / N_SEQ];
      }
      dst[(size_t)b * D + c] = acc;
    }
  }
}

// ---------------------------------------------------------------------------
// f32 -> f16 conversion helpers
// ---------------------------------------------------------------------------
__global__ void cvt_f32_f16(const float* __restrict__ in, _Float16* __restrict__ out, int n) {
  int i = blockIdx.x * 256 + threadIdx.x;
  if (i < n) out[i] = (_Float16)in[i];
}
__global__ void cvt_sum2_f16(const float* __restrict__ a, const float* __restrict__ b,
                             _Float16* __restrict__ out, int n) {
  int i = blockIdx.x * 256 + threadIdx.x;
  if (i < n) out[i] = (_Float16)(a[i] + b[i]);
}

// ---------------------------------------------------------------------------
// Dual-source WMMA GEMM:  C[M,N] = A1[M,K1] @ B1[N,K1]^T + A2[M,K2] @ B2[N,K2]^T
// f16 operands, f32 accumulate. One wave computes a 16x64 tile (4 WMMAs per
// K=32 chunk, A reused 4x). A operand: k = 16*hi + s order == two contiguous
// 16B loads per lane from row-major [M,K]. B operand: same k order (k = 16*hi + s) ==
// one contiguous 32B load per lane from row-major W[N,K] (since C = X @ W^T).
// M % 16 == 0, N % 64 == 0, K % 32 == 0.
// ---------------------------------------------------------------------------
__global__ __launch_bounds__(256)
void dual_gemm_wmma(float* __restrict__ C, int M, int N, int ldc,
                    const _Float16* __restrict__ A1, int lda1,
                    const _Float16* __restrict__ B1, int K1,
                    const _Float16* __restrict__ A2, int lda2,
                    const _Float16* __restrict__ B2, int K2)
{
  const int lane = threadIdx.x & 31;
  const int wave = threadIdx.x >> 5;
  const int hi   = lane >> 4;      // lane group (0: lanes 0-15, 1: lanes 16-31)
  const int l16  = lane & 15;
  const int nTilesN = N >> 6;
  const int gw = blockIdx.x * 8 + wave;
  if (gw >= (M >> 4) * nTilesN) return;
  const int m0 = (gw / nTilesN) << 4;
  const int n0 = (gw % nTilesN) << 6;

  v8f acc0 = {}, acc1 = {}, acc2 = {}, acc3 = {};

  const _Float16* a1row = A1 + (size_t)(m0 + l16) * lda1;
  const _Float16* a2row = A2 + (size_t)(m0 + l16) * lda2;
  const _Float16* b1p0 = B1 + (size_t)(n0 +  0 + l16) * K1;
  const _Float16* b1p1 = B1 + (size_t)(n0 + 16 + l16) * K1;
  const _Float16* b1p2 = B1 + (size_t)(n0 + 32 + l16) * K1;
  const _Float16* b1p3 = B1 + (size_t)(n0 + 48 + l16) * K1;
  const _Float16* b2p0 = B2 + (size_t)(n0 +  0 + l16) * K2;
  const _Float16* b2p1 = B2 + (size_t)(n0 + 16 + l16) * K2;
  const _Float16* b2p2 = B2 + (size_t)(n0 + 32 + l16) * K2;
  const _Float16* b2p3 = B2 + (size_t)(n0 + 48 + l16) * K2;

  for (int k0 = 0; k0 < K1; k0 += 32) {
    v16h a = cat16(*(const v8h*)(a1row + k0 + hi * 16),
                   *(const v8h*)(a1row + k0 + hi * 16 + 8));
    v16h b0 = *(const v16h*)(b1p0 + k0 + hi * 16);
    v16h b1 = *(const v16h*)(b1p1 + k0 + hi * 16);
    v16h b2 = *(const v16h*)(b1p2 + k0 + hi * 16);
    v16h b3 = *(const v16h*)(b1p3 + k0 + hi * 16);
    acc0 = __builtin_amdgcn_wmma_f32_16x16x32_f16(false, a, false, b0, (short)0, acc0, false, false);
    acc1 = __builtin_amdgcn_wmma_f32_16x16x32_f16(false, a, false, b1, (short)0, acc1, false, false);
    acc2 = __builtin_amdgcn_wmma_f32_16x16x32_f16(false, a, false, b2, (short)0, acc2, false, false);
    acc3 = __builtin_amdgcn_wmma_f32_16x16x32_f16(false, a, false, b3, (short)0, acc3, false, false);
  }
  for (int k0 = 0; k0 < K2; k0 += 32) {
    v16h a = cat16(*(const v8h*)(a2row + k0 + hi * 16),
                   *(const v8h*)(a2row + k0 + hi * 16 + 8));
    v16h b0 = *(const v16h*)(b2p0 + k0 + hi * 16);
    v16h b1 = *(const v16h*)(b2p1 + k0 + hi * 16);
    v16h b2 = *(const v16h*)(b2p2 + k0 + hi * 16);
    v16h b3 = *(const v16h*)(b2p3 + k0 + hi * 16);
    acc0 = __builtin_amdgcn_wmma_f32_16x16x32_f16(false, a, false, b0, (short)0, acc0, false, false);
    acc1 = __builtin_amdgcn_wmma_f32_16x16x32_f16(false, a, false, b1, (short)0, acc1, false, false);
    acc2 = __builtin_amdgcn_wmma_f32_16x16x32_f16(false, a, false, b2, (short)0, acc2, false, false);
    acc3 = __builtin_amdgcn_wmma_f32_16x16x32_f16(false, a, false, b3, (short)0, acc3, false, false);
  }
  // C/D f32 layout: element r of the v8f lives at row m0 + r + hi*8, col n0 + l16.
  #pragma unroll
  for (int r = 0; r < 8; ++r) {
    float* crow = C + (size_t)(m0 + hi * 8 + r) * ldc + n0 + l16;
    crow[0]  = acc0[r];
    crow[16] = acc1[r];
    crow[32] = acc2[r];
    crow[48] = acc3[r];
  }
}

// ---------------------------------------------------------------------------
// shift = LayerNorm(f_cross_b + textG) -> f16 for the LSTM GEMM A operand.
// One block per (b, t) row; fcross bias/scale folded in here.
// ---------------------------------------------------------------------------
__global__ __launch_bounds__(256)
void ln_shift_kernel(const float* __restrict__ fcbuf, const float* __restrict__ vb,
                     const float* __restrict__ textG, const float* __restrict__ g,
                     const float* __restrict__ beta, _Float16* __restrict__ shiftH)
{
  const int row = blockIdx.x;          // b*50 + t
  const int b = row / N_SEQ;
  const int tid = threadIdx.x;
  __shared__ float red[256];
  __shared__ float red2[256];
  float x[3], s = 0.f, s2 = 0.f;
  #pragma unroll
  for (int i = 0; i < 3; ++i) {
    int c = tid + i * 256;
    float v = fcbuf[(size_t)b * D + c] * (1.f / 3.f)
            + (vb[c] + vb[D + c] + vb[2 * D + c]) * ((float)N_SEQ / 3.f)
            + textG[(size_t)row * D + c];
    x[i] = v; s += v; s2 += v * v;
  }
  red[tid] = s; red2[tid] = s2; __syncthreads();
  for (int off = 128; off; off >>= 1) {
    if (tid < off) { red[tid] += red[tid + off]; red2[tid] += red2[tid + off]; }
    __syncthreads();
  }
  const float mean = red[0] * (1.f / D);
  const float var  = red2[0] * (1.f / D) - mean * mean;
  const float rstd = rsqrtf(var + 1e-5f);
  #pragma unroll
  for (int i = 0; i < 3; ++i) {
    int c = tid + i * 256;
    shiftH[(size_t)row * D + c] = (_Float16)((x[i] - mean) * rstd * g[c] + beta[c]);
  }
}

// ---------------------------------------------------------------------------
// LSTM pointwise gate update for step t; writes h back as f16 for the next
// step's GEMM and the final residual output out = h + textG directly.
// ---------------------------------------------------------------------------
__global__ __launch_bounds__(256)
void lstm_point_kernel(const float* __restrict__ gbuf, const float* __restrict__ bih,
                       const float* __restrict__ bhh, float* __restrict__ cSt,
                       _Float16* __restrict__ hH, const float* __restrict__ textG,
                       float* __restrict__ out, int t)
{
  const int idx = blockIdx.x * 256 + threadIdx.x;   // B*D
  const int b = idx / D, d = idx - b * D;
  const float* gr = gbuf + (size_t)b * G4;
  float ig = gr[d]          + bih[d]          + bhh[d];
  float fg = gr[D + d]      + bih[D + d]      + bhh[D + d];
  float gg = gr[2 * D + d]  + bih[2 * D + d]  + bhh[2 * D + d];
  float og = gr[3 * D + d]  + bih[3 * D + d]  + bhh[3 * D + d];
  float c = cSt[idx];
  c = sigm(fg) * c + sigm(ig) * tanhf(gg);
  float h = sigm(og) * tanhf(c);
  cSt[idx] = c;
  hH[idx] = (_Float16)h;
  const size_t o = ((size_t)b * N_SEQ + t) * D + d;
  out[o] = h + textG[o];
}

// ---------------------------------------------------------------------------
extern "C" void kernel_launch(void* const* d_in, const int* in_sizes, int n_in,
                              void* d_out, int out_size, void* d_ws, size_t ws_size,
                              hipStream_t stream) {
  const float* te   = (const float*)d_in[0];
  const int*   vid  = (const int*)d_in[1];
  const int*   aid  = (const int*)d_in[2];
  const float* vemb = (const float*)d_in[3];
  const float* aemb = (const float*)d_in[4];
  const float* w3   = (const float*)d_in[5];
  const float* Wih  = (const float*)d_in[6];
  const float* Whh  = (const float*)d_in[7];
  const float* bih  = (const float*)d_in[8];
  const float* bhh  = (const float*)d_in[9];
  const float* lng  = (const float*)d_in[10];
  const float* lnb  = (const float*)d_in[11];
  const float* vW   = (const float*)d_in[20];   // (3, D, D)
  const float* vb   = (const float*)d_in[21];   // (3, D)
  float* out = (float*)d_out;

  // workspace carve-out (256B aligned, ~143 MB total)
  char* ws = (char*)d_ws;
  size_t off = 0;
  auto alloc = [&](size_t bytes) -> char* {
    char* p = ws + off; off = (off + bytes + 255) & ~(size_t)255; return p;
  };
  float*    textG   = (float*)   alloc((size_t)B_SZ * FLAT * 4);
  float*    visSum  = (float*)   alloc((size_t)B_SZ * D * 4);
  float*    acSum   = (float*)   alloc((size_t)B_SZ * D * 4);
  _Float16* visSumH = (_Float16*)alloc((size_t)B_SZ * D * 2);
  _Float16* acSumH  = (_Float16*)alloc((size_t)B_SZ * D * 2);
  _Float16* vW0h    = (_Float16*)alloc((size_t)D * D * 2);
  _Float16* vW12h   = (_Float16*)alloc((size_t)D * D * 2);
  _Float16* WihH    = (_Float16*)alloc((size_t)G4 * D * 2);
  _Float16* WhhH    = (_Float16*)alloc((size_t)G4 * D * 2);
  float*    fcbuf   = (float*)   alloc((size_t)B_SZ * D * 4);
  _Float16* shiftH  = (_Float16*)alloc((size_t)B_SZ * FLAT * 2);
  float*    gbuf    = (float*)   alloc((size_t)B_SZ * G4 * 4);
  _Float16* hH      = (_Float16*)alloc((size_t)B_SZ * D * 2);
  float*    cSt     = (float*)   alloc((size_t)B_SZ * D * 4);

  // 1) EGCE gating: gated text tensor + per-batch column sums for vis/ac
  egce_kernel<<<dim3(B_SZ, 3), 256, 0, stream>>>(te, vid, aid, vemb, aemb, w3,
                                                 textG, visSum, acSum);

  // 2) f16 conversions for WMMA operands
  int n1 = B_SZ * D;
  cvt_f32_f16<<<(n1 + 255) / 256, 256, 0, stream>>>(visSum, visSumH, n1);
  cvt_f32_f16<<<(n1 + 255) / 256, 256, 0, stream>>>(acSum, acSumH, n1);
  int n2 = D * D;
  cvt_f32_f16 <<<(n2 + 255) / 256, 256, 0, stream>>>(vW, vW0h, n2);
  cvt_sum2_f16<<<(n2 + 255) / 256, 256, 0, stream>>>(vW + n2, vW + 2 * n2, vW12h, n2);
  int n3 = G4 * D;
  cvt_f32_f16<<<(n3 + 255) / 256, 256, 0, stream>>>(Wih, WihH, n3);
  cvt_f32_f16<<<(n3 + 255) / 256, 256, 0, stream>>>(Whh, WhhH, n3);

  // 3) collapsed cross-attention: fcbuf = visSum@vW0^T + acSum@(vW1+vW2)^T
  //    (bias + /3 folded into the LN kernel)
  {
    int waves = (B_SZ / 16) * (D / 64);          // 384
    dual_gemm_wmma<<<(waves + 7) / 8, 256, 0, stream>>>(
        fcbuf, B_SZ, D, D, visSumH, D, vW0h, D, acSumH, D, vW12h, D);
  }

  // 4) shift = LN(f_cross + textG) -> f16
  ln_shift_kernel<<<B_SZ * N_SEQ, 256, 0, stream>>>(fcbuf, vb, textG, lng, lnb, shiftH);

  // 5) LSTM: zero state, then 50 sequential WMMA GEMM + pointwise steps
  hipMemsetAsync(hH, 0, (size_t)B_SZ * D * 2, stream);
  hipMemsetAsync(cSt, 0, (size_t)B_SZ * D * 4, stream);
  const int lstmWaves = (B_SZ / 16) * (G4 / 64);  // 1536 -> 192 blocks
  for (int t = 0; t < N_SEQ; ++t) {
    // gbuf = shift[:,t,:] @ Wih^T + h @ Whh^T  (K = 768 + 768)
    dual_gemm_wmma<<<(lstmWaves + 7) / 8, 256, 0, stream>>>(
        gbuf, B_SZ, G4, G4,
        shiftH + (size_t)t * D, N_SEQ * D, WihH, D,
        hH, D, WhhH, D);
    lstm_point_kernel<<<(B_SZ * D) / 256, 256, 0, stream>>>(
        gbuf, bih, bhh, cSt, hH, textG, out, t);
  }
}
